// GNNFiLM_84765474554364
// MI455X (gfx1250) — matrix operand order repaired and checked by reading, not compiled, not run on hardware
//
#include <hip/hip_runtime.h>

typedef unsigned short u16;
typedef __attribute__((ext_vector_type(16))) __bf16 v16bf;
typedef __attribute__((ext_vector_type(8)))  float  v8f;

union Frag16 { v16bf v; uint4 q[2]; };

__device__ __forceinline__ u16 f2bf(float x) {
    unsigned int u = __float_as_uint(x);
    unsigned int r = u + 0x7FFFu + ((u >> 16) & 1u);   // round-to-nearest-even
    return (u16)(r >> 16);
}

// ---------------- conversion / zero utilities ----------------

__global__ void cvt_bf16_kernel(const float* __restrict__ in, u16* __restrict__ out, int n) {
    int i = blockIdx.x * blockDim.x + threadIdx.x;
    if (i < n) out[i] = f2bf(in[i]);
}

__global__ void zero_kernel(float* __restrict__ p, int n) {
    int i = blockIdx.x * blockDim.x + threadIdx.x;
    if (i < n) p[i] = 0.0f;
}

// ---------------- WMMA fragment loaders ----------------
// A: 16x32 bf16 tile of X (row-major, 128 cols). lane<16: K {c*32+0..7, +16..23};
// lane>=16: K {c*32+8..15, +24..31}. Two 16B loads per chunk.
__device__ __forceinline__ void load_a_frags(const u16* __restrict__ X, int row, int lane,
                                             Frag16 a[4]) {
    const uint4* xp = reinterpret_cast<const uint4*>(
        X + (size_t)row * 128 + ((lane & 16) ? 8 : 0));
#pragma unroll
    for (int c = 0; c < 4; ++c) {
        a[c].q[0] = xp[c * 4 + 0];   // +c*32 elems
        a[c].q[1] = xp[c * 4 + 2];   // +16 elems further
    }
}

// B: 32x16 bf16, B[k][o] = Wt[o][k]. lane<16 -> col o=base+lane, K={c*32+0..7, +16..23};
// lane>=16 -> col o=base+lane-16, K={c*32+8..15, +24..31}. Two 16B loads per lane.
__device__ __forceinline__ v16bf load_b_frag(const u16* __restrict__ Wt, int o, int c, int lane) {
    Frag16 f;
    const uint4* p = reinterpret_cast<const uint4*>(
        Wt + (size_t)o * 128 + c * 32 + ((lane & 16) ? 8 : 0));
    f.q[0] = p[0];
    f.q[1] = p[2];
    return f.v;
}

// ---------------- fused dual-GEMM + FiLM + ReLU ----------------
// msg[n,:] = relu(fw[:, :128] * (x@W^T) + fw[:, 128:])  with fw = x@F^T
__global__ void __launch_bounds__(256)
film_gemm_kernel(const u16* __restrict__ X, const u16* __restrict__ Wb,
                 const u16* __restrict__ Fb, float* __restrict__ msg, int ntiles) {
    const int lane = threadIdx.x & 31;
    const int wave = threadIdx.x >> 5;
    const int tile = blockIdx.x * 8 + wave;
    if (tile >= ntiles) return;                 // uniform per wave: EXEC stays all-1s inside
    const int nodeBase = tile * 16;
    const int row = nodeBase + (lane & 15);

    Frag16 a[4];
    load_a_frags(X, row, lane, a);

    const int mrow = nodeBase + ((lane & 16) ? 8 : 0);
    for (int t = 0; t < 8; ++t) {
        const int colBase = t * 16;
        const int o = colBase + (lane & 15);
        v8f am = {}, ag = {}, ab = {};
#pragma unroll
        for (int c = 0; c < 4; ++c) {
            v16bf bm = load_b_frag(Wb, o, c, lane);
            v16bf bg = load_b_frag(Fb, o, c, lane);
            v16bf bb = load_b_frag(Fb, o + 128, c, lane);
            am = __builtin_amdgcn_wmma_f32_16x16x32_bf16(false, a[c].v, false, bm,
                                                         (short)0, am, false, false);
            ag = __builtin_amdgcn_wmma_f32_16x16x32_bf16(false, a[c].v, false, bg,
                                                         (short)0, ag, false, false);
            ab = __builtin_amdgcn_wmma_f32_16x16x32_bf16(false, a[c].v, false, bb,
                                                         (short)0, ab, false, false);
        }
        float* op = msg + (size_t)mrow * 128 + colBase + (lane & 15);
#pragma unroll
        for (int j = 0; j < 8; ++j) {
            float r = ag[j] * am[j] + ab[j];
            op[(size_t)j * 128] = r > 0.0f ? r : 0.0f;
        }
    }
}

// ---------------- edge scatter-add (one wave per edge) ----------------
__global__ void __launch_bounds__(256)
scatter_kernel(const float* __restrict__ msg, const int* __restrict__ src,
               const int* __restrict__ dst, float* __restrict__ h, int e) {
    const int wid = (blockIdx.x * blockDim.x + threadIdx.x) >> 5;
    const int lane = threadIdx.x & 31;
    if (wid >= e) return;
    const int s = src[wid];
    const int d = dst[wid];
    const float4 v = reinterpret_cast<const float4*>(msg + (size_t)s * 128)[lane];
    float* hp = h + (size_t)d * 128 + lane * 4;
    __hip_atomic_fetch_add(hp + 0, v.x, __ATOMIC_RELAXED, __HIP_MEMORY_SCOPE_AGENT);
    __hip_atomic_fetch_add(hp + 1, v.y, __ATOMIC_RELAXED, __HIP_MEMORY_SCOPE_AGENT);
    __hip_atomic_fetch_add(hp + 2, v.z, __ATOMIC_RELAXED, __HIP_MEMORY_SCOPE_AGENT);
    __hip_atomic_fetch_add(hp + 3, v.w, __ATOMIC_RELAXED, __HIP_MEMORY_SCOPE_AGENT);
}

// ---------------- fused LayerNorm -> bf16 (wave per node, wave32 shuffles) ----------------
__global__ void __launch_bounds__(256)
ln_bf16_kernel(const float* __restrict__ h, const float* __restrict__ g,
               const float* __restrict__ b, u16* __restrict__ xb, int n) {
    const int wid = (blockIdx.x * blockDim.x + threadIdx.x) >> 5;
    const int lane = threadIdx.x & 31;
    if (wid >= n) return;
    const float4 v = reinterpret_cast<const float4*>(h + (size_t)wid * 128)[lane];
    float s1 = v.x + v.y + v.z + v.w;
    float s2 = v.x * v.x + v.y * v.y + v.z * v.z + v.w * v.w;
#pragma unroll
    for (int off = 16; off > 0; off >>= 1) {
        s1 += __shfl_xor(s1, off, 32);
        s2 += __shfl_xor(s2, off, 32);
    }
    const float mu = s1 * (1.0f / 128.0f);
    const float var = s2 * (1.0f / 128.0f) - mu * mu;
    const float rstd = rsqrtf(var + 1e-5f);
    const float4 gv = reinterpret_cast<const float4*>(g)[lane];
    const float4 bv = reinterpret_cast<const float4*>(b)[lane];
    ushort4 o;
    o.x = f2bf((v.x - mu) * rstd * gv.x + bv.x);
    o.y = f2bf((v.y - mu) * rstd * gv.y + bv.y);
    o.z = f2bf((v.z - mu) * rstd * gv.z + bv.z);
    o.w = f2bf((v.w - mu) * rstd * gv.w + bv.w);
    reinterpret_cast<ushort4*>(xb + (size_t)wid * 128)[lane] = o;
}

// ---------------- final projection + sigmoid (WMMA) ----------------
__global__ void __launch_bounds__(256)
proj_kernel(const u16* __restrict__ X, const u16* __restrict__ Wpb,
            const float* __restrict__ bp, float* __restrict__ out, int ntiles) {
    const int lane = threadIdx.x & 31;
    const int wave = threadIdx.x >> 5;
    const int tile = blockIdx.x * 8 + wave;
    if (tile >= ntiles) return;
    const int nodeBase = tile * 16;
    const int row = nodeBase + (lane & 15);

    Frag16 a[4];
    load_a_frags(X, row, lane, a);

    const int mrow = nodeBase + ((lane & 16) ? 8 : 0);
    for (int t = 0; t < 4; ++t) {
        const int colBase = t * 16;
        const int o = colBase + (lane & 15);
        v8f acc = {};
#pragma unroll
        for (int c = 0; c < 4; ++c) {
            v16bf bw = load_b_frag(Wpb, o, c, lane);
            acc = __builtin_amdgcn_wmma_f32_16x16x32_bf16(false, a[c].v, false, bw,
                                                          (short)0, acc, false, false);
        }
        const float bias = bp[colBase + (lane & 15)];
        float* op = out + (size_t)mrow * 64 + colBase + (lane & 15);
#pragma unroll
        for (int j = 0; j < 8; ++j) {
            float x = acc[j] + bias;
            op[(size_t)j * 64] = 1.0f / (1.0f + __expf(-x));
        }
    }
}

// ---------------- launch ----------------

extern "C" void kernel_launch(void* const* d_in, const int* in_sizes, int n_in,
                              void* d_out, int out_size, void* d_ws, size_t ws_size,
                              hipStream_t stream) {
    const float* features = (const float*)d_in[0];
    const int*   src      = (const int*)d_in[1];
    const int*   dst      = (const int*)d_in[2];
    const float* W1       = (const float*)d_in[3];
    const float* F1       = (const float*)d_in[4];
    const float* g1       = (const float*)d_in[5];
    const float* b1       = (const float*)d_in[6];
    const float* W2       = (const float*)d_in[7];
    const float* F2       = (const float*)d_in[8];
    const float* g2       = (const float*)d_in[9];
    const float* b2       = (const float*)d_in[10];
    const float* Wp       = (const float*)d_in[11];
    const float* bp       = (const float*)d_in[12];

    const int N = in_sizes[0] / 128;           // 50000
    const int E = in_sizes[1];                 // 600000
    const int NF = N * 128;
    const int ntiles = N / 16;                 // 3125 (N is a multiple of 16)

    char* ws   = (char*)d_ws;
    u16*  xb   = (u16*)ws;                                   // N*128 bf16  (12.8 MB)
    float* msg = (float*)(ws + (size_t)NF * 2);              // N*128 f32   (25.6 MB)
    float* h   = (float*)(ws + (size_t)NF * 2 + (size_t)NF * 4); // N*128 f32
    u16*  W1b  = (u16*)(ws + (size_t)NF * 2 + (size_t)NF * 8);
    u16*  F1b  = W1b + 128 * 128;
    u16*  W2b  = F1b + 256 * 128;
    u16*  F2b  = W2b + 128 * 128;
    u16*  Wpb  = F2b + 256 * 128;

    const int cvtBlocks = (NF + 255) / 256;
    const int gemmBlocks = (ntiles + 7) / 8;
    const int edgeBlocks = (E + 7) / 8;        // 8 waves (edges) per 256-thread block
    const int lnBlocks = (N + 7) / 8;

    // bf16 conversions (weights + features)
    cvt_bf16_kernel<<<cvtBlocks, 256, 0, stream>>>(features, xb, NF);
    cvt_bf16_kernel<<<64, 256, 0, stream>>>(W1, W1b, 128 * 128);
    cvt_bf16_kernel<<<128, 256, 0, stream>>>(F1, F1b, 256 * 128);
    cvt_bf16_kernel<<<64, 256, 0, stream>>>(W2, W2b, 128 * 128);
    cvt_bf16_kernel<<<128, 256, 0, stream>>>(F2, F2b, 256 * 128);
    cvt_bf16_kernel<<<32, 256, 0, stream>>>(Wp, Wpb, 64 * 128);

    // ---- layer 1 ----
    zero_kernel<<<cvtBlocks, 256, 0, stream>>>(h, NF);
    film_gemm_kernel<<<gemmBlocks, 256, 0, stream>>>(xb, W1b, F1b, msg, ntiles);
    scatter_kernel<<<edgeBlocks, 256, 0, stream>>>(msg, src, dst, h, E);
    ln_bf16_kernel<<<lnBlocks, 256, 0, stream>>>(h, g1, b1, xb, N);

    // ---- layer 2 ----
    zero_kernel<<<cvtBlocks, 256, 0, stream>>>(h, NF);
    film_gemm_kernel<<<gemmBlocks, 256, 0, stream>>>(xb, W2b, F2b, msg, ntiles);
    scatter_kernel<<<edgeBlocks, 256, 0, stream>>>(msg, src, dst, h, E);
    ln_bf16_kernel<<<lnBlocks, 256, 0, stream>>>(h, g2, b2, xb, N);

    // ---- projection + sigmoid ----
    proj_kernel<<<gemmBlocks, 256, 0, stream>>>(xb, Wpb, bp, (float*)d_out, ntiles);
}
